// force_block_68427418960433
// MI455X (gfx1250) — matrix operand order repaired and checked by reading, not compiled, not run on hardware
//
#include <hip/hip_runtime.h>

// ---------------------------------------------------------------------------
// MI455X (gfx1250) edge-MLP force block.
// Heavy op: feats[E,262] @ W1[262,128]  (~54 GFLOP fp32-equivalent).
// Node tables fit in 192MB L2 -> compute bound -> use v_wmma_f32_16x16x32_bf16
// with split-bf16 (hi/lo) error compensation (3 products) for ~fp32 accuracy.
// ---------------------------------------------------------------------------

#define HDIM 128
#define KIN  262            // 2H + 6
#define KP   288            // padded to 9 * 32
#define KSTEPS 9
#define EPSN 1e-30f
#define EPSLN 1e-5f

typedef __attribute__((ext_vector_type(16))) __bf16 v16bf;
typedef __attribute__((ext_vector_type(8)))  __bf16 v8bf;
typedef __attribute__((ext_vector_type(8)))  float  v8f;
typedef __attribute__((ext_vector_type(4)))  float  v4f;
typedef __attribute__((ext_vector_type(4)))  unsigned int v4u;

// LDS layout (bytes), dynamic shared memory
#define SM_W1HI 0
#define SM_W1LO (HDIM * KP * 2)               // 73728
#define SM_A    (2 * HDIM * KP * 2)           // 147456
#define SM_AHI(w) (SM_A + (w) * 16 * KP * 2)
#define SM_ALO(w) (SM_A + 8 * 16 * KP * 2 + (w) * 16 * KP * 2)  // 221184 + ...
#define SM_FR(w)  (SM_A + 16 * 16 * KP * 2 + (w) * 16 * 9 * 4)  // 294912 + ...
#define SM_TOTAL  (SM_A + 16 * 16 * KP * 2 + 8 * 16 * 9 * 4)    // 299520 < 320K

__device__ __forceinline__ unsigned short f2bf_rn(float f) {
  unsigned u = __float_as_uint(f);
  u += 0x7FFFu + ((u >> 16) & 1u);       // round-to-nearest-even
  return (unsigned short)(u >> 16);
}
__device__ __forceinline__ float bf2f(unsigned short h) {
  return __uint_as_float(((unsigned)h) << 16);
}

__device__ __forceinline__ v8f wmma_bf16(v16bf a, v16bf b, v8f c) {
  return __builtin_amdgcn_wmma_f32_16x16x32_bf16(false, a, false, b,
                                                 (short)0, c, false, false);
}

// ---------------------------------------------------------------------------
// Prep: W1[262][128] f32 -> W1T[128][288] bf16 hi + lo (transposed, K-padded)
// ---------------------------------------------------------------------------
__global__ void fb_prep_w1(const float* __restrict__ W1,
                           unsigned short* __restrict__ w1t) {
  int idx = blockIdx.x * blockDim.x + threadIdx.x;
  if (idx >= HDIM * KP) return;
  int n = idx / KP, kp = idx % KP;
  float v = (kp < KIN) ? W1[kp * HDIM + n] : 0.0f;
  unsigned short hv = f2bf_rn(v);
  float r = v - bf2f(hv);
  w1t[idx] = hv;                    // hi table
  w1t[HDIM * KP + idx] = f2bf_rn(r); // lo table (contiguous after hi)
}

__global__ void fb_zero(float* __restrict__ p, int n) {
  int i = blockIdx.x * blockDim.x + threadIdx.x;
  if (i < n) p[i] = 0.0f;
}

// ---------------------------------------------------------------------------
// Main: per 16-edge wave tile, GEMM via 216 bf16 WMMAs (hi/lo split).
// ---------------------------------------------------------------------------
__global__ void __launch_bounds__(256)
fb_main(const float* __restrict__ x, const float* __restrict__ h,
        const int* __restrict__ ei, const unsigned short* __restrict__ w1t,
        const float* __restrict__ b1, const float* __restrict__ gam_g,
        const float* __restrict__ bet_g, const float* __restrict__ W2,
        float* __restrict__ sums, float* __restrict__ cnts, int Nn, int Ee) {
  extern __shared__ char smem[];
  const int wave = threadIdx.x >> 5;
  const int lane = threadIdx.x & 31;
  const int mrow = lane & 15;
  const int half = lane >> 4;

  unsigned short* sW1hi = (unsigned short*)(smem + SM_W1HI);
  unsigned short* sW1lo = (unsigned short*)(smem + SM_W1LO);
  unsigned short* sAhi  = (unsigned short*)(smem + SM_AHI(wave));
  unsigned short* sAlo  = (unsigned short*)(smem + SM_ALO(wave));
  float*          sFr   = (float*)(smem + SM_FR(wave));

  // --- cooperative W1T (hi+lo) load into LDS: 147456 B = 9216 x 16B ---
  {
    const v4u* src = (const v4u*)w1t;
    v4u* dst = (v4u*)smem;
    for (int i = threadIdx.x; i < (2 * HDIM * KP * 2) / 16; i += 256)
      dst[i] = src[i];
  }
  __syncthreads();

  const int tiles = (Ee + 127) >> 7;   // 128 edges per block-iteration
  for (int t = blockIdx.x; t < tiles; t += gridDim.x) {
    const int e0 = (t << 7) + (wave << 4);   // this wave's 16 edges

    // ---------------- fill phase ----------------
    if (lane < 16) {                 // geometry for edge m = lane
      int e = e0 + lane; if (e >= Ee) e = Ee - 1;
      int r = ei[e], c = ei[Ee + e];
      float xi0 = x[3*r], xi1 = x[3*r+1], xi2 = x[3*r+2];
      float xj0 = x[3*c], xj1 = x[3*c+1], xj2 = x[3*c+2];
      float d0 = xi0-xj0, d1 = xi1-xj1, d2 = xi2-xj2;
      float c0 = xi1*xj2 - xi2*xj1;
      float c1 = xi2*xj0 - xi0*xj2;
      float c2 = xi0*xj1 - xi1*xj0;
      float dn = 1.0f / sqrtf(d0*d0 + d1*d1 + d2*d2 + 3.0f*EPSN);
      d0 *= dn; d1 *= dn; d2 *= dn;
      float cn = 1.0f / sqrtf(c0*c0 + c1*c1 + c2*c2 + 3.0f*EPSN);
      c0 *= cn; c1 *= cn; c2 *= cn;
      float v0 = d1*c2 - d2*c1, v1 = d2*c0 - d0*c2, v2 = d0*c1 - d1*c0;
      float* fr = sFr + lane * 9;
      fr[0]=d0; fr[1]=d1; fr[2]=d2; fr[3]=c0; fr[4]=c1; fr[5]=c2;
      fr[6]=v0; fr[7]=v1; fr[8]=v2;
      float feat[6];
      feat[0] = xi0*d0 + xi1*d1 + xi2*d2;
      feat[1] = xi0*c0 + xi1*c1 + xi2*c2;
      feat[2] = xi0*v0 + xi1*v1 + xi2*v2;
      feat[3] = xj0*d0 + xj1*d1 + xj2*d2;
      feat[4] = xj0*c0 + xj1*c1 + xj2*c2;
      feat[5] = xj0*v0 + xj1*v1 + xj2*v2;
      #pragma unroll
      for (int k = 0; k < 32; ++k) {        // features 256..261, pad to 287
        float val = (k < 6) ? feat[k] : 0.0f;
        unsigned short hv = f2bf_rn(val);
        sAhi[lane * KP + 256 + k] = hv;
        sAlo[lane * KP + 256 + k] = f2bf_rn(val - bf2f(hv));
      }
    }
    {   // hidden-state gather: lane = (edge m, row/col half)
      int e = e0 + mrow; if (e >= Ee) e = Ee - 1;
      int node = ei[half * Ee + e];
      const v4f* hp = (const v4f*)(h + (size_t)node * HDIM);
      unsigned int* dh = (unsigned int*)(sAhi + mrow * KP + half * HDIM);
      unsigned int* dl = (unsigned int*)(sAlo + mrow * KP + half * HDIM);
      #pragma unroll 8
      for (int i = 0; i < 32; ++i) {
        v4f vv = hp[i];
        unsigned short a0 = f2bf_rn(vv[0]); float r0 = vv[0] - bf2f(a0);
        unsigned short a1 = f2bf_rn(vv[1]); float r1 = vv[1] - bf2f(a1);
        unsigned short a2 = f2bf_rn(vv[2]); float r2 = vv[2] - bf2f(a2);
        unsigned short a3 = f2bf_rn(vv[3]); float r3 = vv[3] - bf2f(a3);
        dh[2*i]   = (unsigned)a0 | ((unsigned)a1 << 16);
        dh[2*i+1] = (unsigned)a2 | ((unsigned)a3 << 16);
        dl[2*i]   = (unsigned)f2bf_rn(r0) | ((unsigned)f2bf_rn(r1) << 16);
        dl[2*i+1] = (unsigned)f2bf_rn(r2) | ((unsigned)f2bf_rn(r3) << 16);
      }
    }
    __syncthreads();

    // ---------------- GEMM: z[16,128] = A[16,288] @ W1T^T ----------------
    v8f acc[8] = {};
    auto ldA = [&](const unsigned short* base, int s) -> v16bf {
      int o = mrow * KP + 32 * s + ((lane & 16) ? 8 : 0);
      v8bf p0 = *(const v8bf*)(base + o);       // K chunk [0..7] (+8 hi-half)
      v8bf p1 = *(const v8bf*)(base + o + 16);  // K chunk [16..23] (+8)
      return __builtin_shufflevector(p0, p1, 0,1,2,3,4,5,6,7,
                                             8,9,10,11,12,13,14,15);
    };
    auto ldB = [&](const unsigned short* base, int s, int n) -> v16bf {
      int o = (16 * n + mrow) * KP + 32 * s + ((lane & 16) ? 8 : 0);
      v8bf p0 = *(const v8bf*)(base + o);       // K chunk [0..7] (+8 hi-half)
      v8bf p1 = *(const v8bf*)(base + o + 16);  // K chunk [16..23] (+8)
      return __builtin_shufflevector(p0, p1, 0,1,2,3,4,5,6,7,
                                             8,9,10,11,12,13,14,15);
    };
    #pragma unroll
    for (int s = 0; s < KSTEPS; ++s) {
      v16bf ah = ldA(sAhi, s);
      v16bf al = ldA(sAlo, s);
      #pragma unroll
      for (int np = 0; np < 4; ++np) {
        const int n0 = 2 * np, n1 = n0 + 1;
        v16bf bh0 = ldB(sW1hi, s, n0), bh1 = ldB(sW1hi, s, n1);
        v16bf bl0 = ldB(sW1lo, s, n0), bl1 = ldB(sW1lo, s, n1);
        acc[n0] = wmma_bf16(ah, bh0, acc[n0]);
        acc[n1] = wmma_bf16(ah, bh1, acc[n1]);
        acc[n0] = wmma_bf16(al, bh0, acc[n0]);
        acc[n1] = wmma_bf16(al, bh1, acc[n1]);
        acc[n0] = wmma_bf16(ah, bl0, acc[n0]);
        acc[n1] = wmma_bf16(ah, bl1, acc[n1]);
      }
    }

    // ---------------- epilogue: bias + SiLU ----------------
    float biasv[8], gamv[8], betv[8];
    #pragma unroll
    for (int n = 0; n < 8; ++n) {
      int N = 16 * n + mrow;
      biasv[n] = b1[N]; gamv[n] = gam_g[N]; betv[n] = bet_g[N];
    }
    #pragma unroll
    for (int n = 0; n < 8; ++n) {
      #pragma unroll
      for (int v = 0; v < 8; ++v) {
        float z = acc[n][v] + biasv[n];
        acc[n][v] = z / (1.0f + __expf(-z));
      }
    }
    // LayerNorm stats per row (row M = v + 8*half spans 16 lanes of the half)
    float muv[8], rsd[8];
    #pragma unroll
    for (int v = 0; v < 8; ++v) {
      float s1 = 0.0f, s2 = 0.0f;
      #pragma unroll
      for (int n = 0; n < 8; ++n) { float z = acc[n][v]; s1 += z; s2 += z*z; }
      #pragma unroll
      for (int mk = 1; mk <= 8; mk <<= 1) {
        s1 += __shfl_xor(s1, mk, 32);
        s2 += __shfl_xor(s2, mk, 32);
      }
      float m = s1 * (1.0f / 128.0f);
      float var = s2 * (1.0f / 128.0f) - m * m;
      muv[v] = m; rsd[v] = 1.0f / sqrtf(var + EPSLN);
    }
    // normalize, z @ W2 -> invariant force (reduced across the 16-lane half)
    float w2v[8][3];
    #pragma unroll
    for (int n = 0; n < 8; ++n) {
      int N = 16 * n + mrow;
      w2v[n][0] = W2[3*N]; w2v[n][1] = W2[3*N+1]; w2v[n][2] = W2[3*N+2];
    }
    float ff[8][3];
    #pragma unroll
    for (int v = 0; v < 8; ++v) {
      float f0 = 0.0f, f1 = 0.0f, f2 = 0.0f;
      #pragma unroll
      for (int n = 0; n < 8; ++n) {
        float z = (acc[n][v] - muv[v]) * rsd[v] * gamv[n] + betv[n];
        f0 += z * w2v[n][0]; f1 += z * w2v[n][1]; f2 += z * w2v[n][2];
      }
      #pragma unroll
      for (int mk = 1; mk <= 8; mk <<= 1) {
        f0 += __shfl_xor(f0, mk, 32);
        f1 += __shfl_xor(f1, mk, 32);
        f2 += __shfl_xor(f2, mk, 32);
      }
      ff[v][0] = f0; ff[v][1] = f1; ff[v][2] = f2;
    }
    // rotate back, clip, atomic segment-sum (lanes with (lane&15)==v emit)
    #pragma unroll
    for (int v = 0; v < 8; ++v) {
      if (mrow == v) {
        int m = half * 8 + v;
        int e = e0 + m;
        if (e < Ee) {
          const float* fr = sFr + m * 9;
          float i0 = ff[v][0], i1 = ff[v][1], i2 = ff[v][2];
          float g0 = i0*fr[0] + i1*fr[3] + i2*fr[6];
          float g1 = i0*fr[1] + i1*fr[4] + i2*fr[7];
          float g2 = i0*fr[2] + i1*fr[5] + i2*fr[8];
          g0 = fminf(fmaxf(g0, -1000.0f), 1000.0f);
          g1 = fminf(fmaxf(g1, -1000.0f), 1000.0f);
          g2 = fminf(fmaxf(g2, -1000.0f), 1000.0f);
          int r = ei[e];
          atomicAdd(&sums[3*r+0], g0);
          atomicAdd(&sums[3*r+1], g1);
          atomicAdd(&sums[3*r+2], g2);
          atomicAdd(&cnts[r], 1.0f);
        }
      }
    }
    __syncthreads();   // protect per-wave A staging before next fill
  }
}

__global__ void fb_finalize(const float* __restrict__ sums,
                            const float* __restrict__ cnts,
                            float* __restrict__ out, int Nn) {
  int i = blockIdx.x * blockDim.x + threadIdx.x;
  if (i >= Nn * 3) return;
  float c = cnts[i / 3];
  out[i] = sums[i] / ((c == 0.0f) ? 1.0f : c);
}

// ---------------------------------------------------------------------------
extern "C" void kernel_launch(void* const* d_in, const int* in_sizes, int n_in,
                              void* d_out, int out_size, void* d_ws, size_t ws_size,
                              hipStream_t stream) {
  const float* x  = (const float*)d_in[0];
  const float* h  = (const float*)d_in[1];
  const int*   ei = (const int*)d_in[2];
  const float* W1 = (const float*)d_in[3];
  const float* b1 = (const float*)d_in[4];
  const float* ga = (const float*)d_in[5];
  const float* be = (const float*)d_in[6];
  const float* W2 = (const float*)d_in[7];
  float* out = (float*)d_out;

  const int Nn = in_sizes[0] / 3;
  const int Ee = in_sizes[2] / 2;

  // workspace layout: sums[3N] | cnts[N] | W1T hi+lo (bf16)
  float* sums = (float*)d_ws;
  float* cnts = sums + (size_t)Nn * 3;
  size_t off = ((size_t)Nn * 4 * sizeof(float) + 255) & ~(size_t)255;
  unsigned short* w1t = (unsigned short*)((char*)d_ws + off);

  (void)hipFuncSetAttribute((const void*)fb_main,
                            hipFuncAttributeMaxDynamicSharedMemorySize,
                            SM_TOTAL);

  fb_zero<<<(Nn * 4 + 255) / 256, 256, 0, stream>>>(sums, Nn * 4);
  fb_prep_w1<<<(HDIM * KP + 255) / 256, 256, 0, stream>>>(W1, w1t);

  int tiles = (Ee + 127) >> 7;
  int grid = tiles < 640 ? tiles : 640;
  fb_main<<<grid, 256, SM_TOTAL, stream>>>(x, h, ei, w1t, b1, ga, be, W2,
                                           sums, cnts, Nn, Ee);
  fb_finalize<<<(Nn * 3 + 255) / 256, 256, 0, stream>>>(sums, cnts, out, Nn);
}
